// GINENet_80865644249571
// MI455X (gfx1250) — matrix operand order repaired and checked by reading, not compiled, not run on hardware
//
#include <hip/hip_runtime.h>
#include <hip/hip_bf16.h>
#include <math.h>

typedef __attribute__((ext_vector_type(16))) _Float16 v16h;
typedef __attribute__((ext_vector_type(8)))  float    v8f;

// ---------------------------------------------------------------------------
// order-preserving float<->uint encoding for atomicMax-based segment max
// ---------------------------------------------------------------------------
__device__ __forceinline__ unsigned encf(float f) {
    unsigned u = __float_as_uint(f);
    return (u & 0x80000000u) ? ~u : (u | 0x80000000u);
}
__device__ __forceinline__ float decf(unsigned u) {
    u = (u & 0x80000000u) ? (u & 0x7FFFFFFFu) : ~u;
    return __uint_as_float(u);
}
__device__ __forceinline__ float relu_(float v) { return v > 0.f ? v : 0.f; }
__device__ __forceinline__ float sig_(float v)  { return 1.f / (1.f + expf(-v)); }

// ---------------------------------------------------------------------------
// Edge pass: acc[dst] += relu(x[src] + ea @ ew + eb). acc pre-initialized
// with x so the result is t = x + agg (GINE, eps = 0).
// ---------------------------------------------------------------------------
template <int FO>
__global__ void __launch_bounds__(256)
edge_kernel(const float* __restrict__ ea1, const float* __restrict__ ea2,
            const int*   __restrict__ eidx,
            const float* __restrict__ ew,  const float* __restrict__ eb,
            const float* __restrict__ xin, float* __restrict__ acc_out, int nE)
{
    __shared__ float ws[16 * FO];
    __shared__ float bs[FO];
    for (int i = threadIdx.x; i < 16 * FO; i += 256) ws[i] = ew[i];
    for (int i = threadIdx.x; i < FO;      i += 256) bs[i] = eb[i];
    __syncthreads();

    int e = blockIdx.x * 256 + threadIdx.x;
    if (e >= nE) return;

    float ea[16];
#pragma unroll
    for (int k = 0; k < 8; ++k) ea[k]     = ea1[(long)e * 8 + k];
#pragma unroll
    for (int k = 0; k < 8; ++k) ea[8 + k] = ea2[(long)e * 8 + k];

    int src = eidx[e];
    int dst = eidx[nE + e];
    const float* xr = xin + (long)src * FO;
    float*       ar = acc_out + (long)dst * FO;

#pragma unroll
    for (int f = 0; f < FO; ++f) {
        float m = bs[f];
#pragma unroll
        for (int k = 0; k < 16; ++k) m += ea[k] * ws[k * FO + f];
        m = relu_(m + xr[f]);
        atomicAdd(&ar[f], m);
    }
}

// ---------------------------------------------------------------------------
// Node MLP: hout = relu( relu(t @ w1 + b1) @ w2 + b2 ), w1:[KIN,64] w2:[64,64]
// One wave per 16-row tile; 4 waves / block. WMMA f16 inputs, f32 accum.
// Fragment layouts per CDNA5 ISA 7.12.2 (wave32):
//   A 16x32 f16 : lane(0-15)=row M, lane(16-31)=same M other K half;
//                 half j -> K = (j>>3)*16 + (lane>>4)*8 + (j&7)
//   B 32x16 f16 : lane%16 = col N, lane>>4 selects K half;
//                 half j -> K = (j>>3)*16 + (lane>>4)*8 + (j&7)
//   C/D 16x16 f32: VGPR r -> M = r + 8*(lane>>4), N = lane%16
// ---------------------------------------------------------------------------
template <int KIN>
__global__ void __launch_bounds__(128)
mlp_kernel(const float* __restrict__ tin,
           const float* __restrict__ w1, const float* __restrict__ b1,
           const float* __restrict__ w2, const float* __restrict__ b2,
           float* __restrict__ hout, int n)
{
    constexpr int NC1 = KIN / 32;
    __shared__ float w1s[KIN * 64];
    __shared__ float w2s[64 * 64];
    __shared__ float b1s[64], b2s[64];
    __shared__ float Ut[4][16 * 64];

    int tid = threadIdx.x;
    for (int i = tid; i < KIN * 64; i += 128) w1s[i] = w1[i];
    for (int i = tid; i < 64 * 64;  i += 128) w2s[i] = w2[i];
    if (tid < 64) { b1s[tid] = b1[tid]; b2s[tid] = b2[tid]; }
    __syncthreads();

    int w = tid >> 5, lane = tid & 31;
    int g = lane >> 4, ln = lane & 15;
    int tile = blockIdx.x * 4 + w;
    int row  = tile * 16 + ln;
    bool rok = row < n;
    const float* trow = tin + (long)(rok ? row : 0) * KIN;

    // --- layer 1: A fragments from global t ---
    v16h a[NC1];
#pragma unroll
    for (int kc = 0; kc < NC1; ++kc) {
#pragma unroll
        for (int j = 0; j < 16; ++j) {
            int k = kc * 32 + ((j >> 3) << 4) + (g << 3) + (j & 7);
            float v = rok ? trow[k] : 0.0f;
            a[kc][j] = (_Float16)v;
        }
    }
#pragma unroll
    for (int ct = 0; ct < 4; ++ct) {
        v8f c = {};
#pragma unroll
        for (int kc = 0; kc < NC1; ++kc) {
            v16h bf;
#pragma unroll
            for (int j = 0; j < 16; ++j) {
                int k = kc * 32 + ((j >> 3) << 4) + (g << 3) + (j & 7);
                bf[j] = (_Float16)w1s[k * 64 + ct * 16 + ln];
            }
            c = __builtin_amdgcn_wmma_f32_16x16x32_f16(
                    false, a[kc], false, bf, (short)0, c, false, false);
        }
#pragma unroll
        for (int r = 0; r < 8; ++r) {
            int m = r + (g << 3);
            Ut[w][m * 64 + ct * 16 + ln] = relu_(c[r] + b1s[ct * 16 + ln]);
        }
    }
    __syncthreads();

    // --- layer 2: A fragments from LDS intermediate (K = 64) ---
    v16h a2[2];
#pragma unroll
    for (int kc = 0; kc < 2; ++kc) {
#pragma unroll
        for (int j = 0; j < 16; ++j) {
            int k = kc * 32 + ((j >> 3) << 4) + (g << 3) + (j & 7);
            a2[kc][j] = (_Float16)Ut[w][ln * 64 + k];
        }
    }
#pragma unroll
    for (int ct = 0; ct < 4; ++ct) {
        v8f c = {};
#pragma unroll
        for (int kc = 0; kc < 2; ++kc) {
            v16h bf;
#pragma unroll
            for (int j = 0; j < 16; ++j) {
                int k = kc * 32 + ((j >> 3) << 4) + (g << 3) + (j & 7);
                bf[j] = (_Float16)w2s[k * 64 + ct * 16 + ln];
            }
            c = __builtin_amdgcn_wmma_f32_16x16x32_f16(
                    false, a2[kc], false, bf, (short)0, c, false, false);
        }
#pragma unroll
        for (int r = 0; r < 8; ++r) {
            int m = r + (g << 3);
            int orow = tile * 16 + m;
            if (orow < n)
                hout[(long)orow * 64 + ct * 16 + ln] =
                    relu_(c[r] + b2s[ct * 16 + ln]);
        }
    }
}

// ---------------------------------------------------------------------------
// Set2Set LSTM step (torch gate order i,f,g,o). One block per graph.
// ---------------------------------------------------------------------------
__global__ void __launch_bounds__(64)
lstm_step(const float* __restrict__ wih, const float* __restrict__ bih,
          const float* __restrict__ whh, const float* __restrict__ bhh,
          const float* __restrict__ qstar,
          float* __restrict__ hS, float* __restrict__ cS)
{
    int b = blockIdx.x, d = threadIdx.x;
    __shared__ float qs[128];
    __shared__ float hh[64];
    qs[d]      = qstar[b * 128 + d];
    qs[64 + d] = qstar[b * 128 + 64 + d];
    hh[d]      = hS[b * 64 + d];
    __syncthreads();

    float gate[4];
#pragma unroll
    for (int gi = 0; gi < 4; ++gi) {
        int u = gi * 64 + d;
        float acc = bih[u] + bhh[u];
        const float* wr = wih + (long)u * 128;
        for (int k = 0; k < 128; ++k) acc += qs[k] * wr[k];
        const float* wr2 = whh + (long)u * 64;
        for (int k = 0; k < 64; ++k) acc += hh[k] * wr2[k];
        gate[gi] = acc;
    }
    float c = sig_(gate[1]) * cS[b * 64 + d] + sig_(gate[0]) * tanhf(gate[2]);
    cS[b * 64 + d] = c;
    hS[b * 64 + d] = sig_(gate[3]) * tanhf(c);
}

__global__ void init_att(float* __restrict__ r, float* __restrict__ denom,
                         unsigned* __restrict__ mEnc, int B)
{
    int i = blockIdx.x * 256 + threadIdx.x;
    if (i < B * 64) r[i] = 0.f;
    if (i < B) { denom[i] = 0.f; mEnc[i] = encf(-3.4e38f); }
}

__global__ void att_dot(const float* __restrict__ h1, const int* __restrict__ batch,
                        const float* __restrict__ hS, float* __restrict__ e,
                        unsigned* __restrict__ mEnc, int n)
{
    int i = blockIdx.x * 256 + threadIdx.x;
    if (i >= n) return;
    int b = batch[i];
    const float* q  = hS + (long)b * 64;
    const float* xr = h1 + (long)i * 64;
    float s = 0.f;
#pragma unroll
    for (int d = 0; d < 64; ++d) s += xr[d] * q[d];
    e[i] = s;
    atomicMax(&mEnc[b], encf(s));
}

__global__ void att_exp(const int* __restrict__ batch, const float* __restrict__ e,
                        const unsigned* __restrict__ mEnc,
                        float* __restrict__ denom, float* __restrict__ exv, int n)
{
    int i = blockIdx.x * 256 + threadIdx.x;
    if (i >= n) return;
    int b = batch[i];
    float ex = expf(e[i] - decf(mEnc[b]));
    exv[i] = ex;
    atomicAdd(&denom[b], ex);
}

__global__ void att_r(const float* __restrict__ h1, const int* __restrict__ batch,
                      const float* __restrict__ exv, const float* __restrict__ denom,
                      float* __restrict__ r, int n)
{
    int i = blockIdx.x * 256 + threadIdx.x;
    if (i >= n) return;
    int b = batch[i];
    float a = exv[i] / denom[b];
    const float* xr = h1 + (long)i * 64;
#pragma unroll
    for (int d = 0; d < 64; ++d) atomicAdd(&r[b * 64 + d], a * xr[d]);
}

__global__ void qstar_k(const float* __restrict__ hS, const float* __restrict__ r,
                        float* __restrict__ qstar, int B)
{
    int i = blockIdx.x * 256 + threadIdx.x;
    if (i >= B * 64) return;
    int b = i >> 6, d = i & 63;
    qstar[b * 128 + d]      = hS[i];
    qstar[b * 128 + 64 + d] = r[i];
}

// ---------------------------------------------------------------------------
// Head: out[b] = relu(qstar[b] @ dense_w + dense_b) @ out_w + out_b
// ---------------------------------------------------------------------------
__global__ void __launch_bounds__(64)
head_k(const float* __restrict__ qstar, const float* __restrict__ dw,
       const float* __restrict__ db, const float* __restrict__ ow,
       const float* __restrict__ ob, float* __restrict__ out)
{
    int b = blockIdx.x, d = threadIdx.x;
    __shared__ float qs[128];
    __shared__ float red[64];
    qs[d]      = qstar[b * 128 + d];
    qs[64 + d] = qstar[b * 128 + 64 + d];
    __syncthreads();
    float acc = db[d];
    for (int k = 0; k < 128; ++k) acc += qs[k] * dw[k * 64 + d];
    red[d] = relu_(acc) * ow[d];
    __syncthreads();
    for (int s = 32; s > 0; s >>= 1) {
        if (d < s) red[d] += red[d + s];
        __syncthreads();
    }
    if (d == 0) out[b] = red[0] + ob[0];
}

// ---------------------------------------------------------------------------
extern "C" void kernel_launch(void* const* d_in, const int* in_sizes, int n_in,
                              void* d_out, int out_size, void* d_ws, size_t ws_size,
                              hipStream_t stream)
{
    (void)n_in; (void)ws_size;
    const float* x    = (const float*)d_in[0];
    const float* ea1  = (const float*)d_in[1];
    const float* ea2  = (const float*)d_in[2];
    const int*   eidx = (const int*)d_in[3];
    const int*   batch= (const int*)d_in[4];
    const float* c0ew = (const float*)d_in[5];
    const float* c0eb = (const float*)d_in[6];
    const float* c0w1 = (const float*)d_in[7];
    const float* c0b1 = (const float*)d_in[8];
    const float* c0w2 = (const float*)d_in[9];
    const float* c0b2 = (const float*)d_in[10];
    const float* c1ew = (const float*)d_in[11];
    const float* c1eb = (const float*)d_in[12];
    const float* c1w1 = (const float*)d_in[13];
    const float* c1b1 = (const float*)d_in[14];
    const float* c1w2 = (const float*)d_in[15];
    const float* c1b2 = (const float*)d_in[16];
    const float* wih  = (const float*)d_in[17];  // [256,128]
    const float* bih  = (const float*)d_in[18];
    const float* whh  = (const float*)d_in[19];  // [256,64]
    const float* bhh  = (const float*)d_in[20];
    const float* dw   = (const float*)d_in[21];
    const float* db   = (const float*)d_in[22];
    const float* ow   = (const float*)d_in[23];
    const float* ob   = (const float*)d_in[24];

    const int N = in_sizes[0] / 32;
    const int E = in_sizes[1] / 8;
    const int B = out_size;

    // workspace carve-up (floats)
    float* t0    = (float*)d_ws;                  // N*32  (x + agg0)
    float* h0    = t0 + (size_t)N * 32;           // N*64
    float* t1    = h0 + (size_t)N * 64;           // N*64  (h0 + agg1)
    float* h1    = t1 + (size_t)N * 64;           // N*64
    float* ev    = h1 + (size_t)N * 64;           // N
    float* exv   = ev + (size_t)N;                // N
    float* hS    = exv + (size_t)N;               // B*64
    float* cS    = hS + (size_t)B * 64;           // B*64
    float* rS    = cS + (size_t)B * 64;           // B*64
    float* qstar = rS + (size_t)B * 64;           // B*128
    float* denom = qstar + (size_t)B * 128;       // B
    unsigned* mEnc = (unsigned*)(denom + B);      // B

    const int eb_ = (E + 255) / 256;
    const int nb_ = (N + 255) / 256;
    const int ntiles = (N + 15) / 16;
    const int mb_ = (ntiles + 3) / 4;
    const int bb_ = (B * 64 + 255) / 256;

    // ---- conv0 ----
    hipMemcpyAsync(t0, x, (size_t)N * 32 * sizeof(float),
                   hipMemcpyDeviceToDevice, stream);
    edge_kernel<32><<<eb_, 256, 0, stream>>>(ea1, ea2, eidx, c0ew, c0eb, x, t0, E);
    mlp_kernel<32><<<mb_, 128, 0, stream>>>(t0, c0w1, c0b1, c0w2, c0b2, h0, N);

    // ---- conv1 ----
    hipMemcpyAsync(t1, h0, (size_t)N * 64 * sizeof(float),
                   hipMemcpyDeviceToDevice, stream);
    edge_kernel<64><<<eb_, 256, 0, stream>>>(ea1, ea2, eidx, c1ew, c1eb, h0, t1, E);
    mlp_kernel<64><<<mb_, 128, 0, stream>>>(t1, c1w1, c1b1, c1w2, c1b2, h1, N);

    // ---- Set2Set ----
    hipMemsetAsync(hS, 0, (size_t)B * (64 * 3 + 128) * sizeof(float), stream);
    for (int step = 0; step < 3; ++step) {
        lstm_step<<<B, 64, 0, stream>>>(wih, bih, whh, bhh, qstar, hS, cS);
        init_att<<<bb_, 256, 0, stream>>>(rS, denom, mEnc, B);
        att_dot<<<nb_, 256, 0, stream>>>(h1, batch, hS, ev, mEnc, N);
        att_exp<<<nb_, 256, 0, stream>>>(batch, ev, mEnc, denom, exv, N);
        att_r  <<<nb_, 256, 0, stream>>>(h1, batch, exv, denom, rS, N);
        qstar_k<<<bb_, 256, 0, stream>>>(hS, rS, qstar, B);
    }

    // ---- head ----
    head_k<<<B, 64, 0, stream>>>(qstar, dw, db, ow, ob, (float*)d_out);
}
